// AllGNNrg_25323127177647
// MI455X (gfx1250) — matrix operand order repaired and checked by reading, not compiled, not run on hardware
//
#include <hip/hip_runtime.h>
#include <hip/hip_bf16.h>
#include <math.h>

#define GDIM 64
#define GHEAD 8
#define GDH 8

typedef __attribute__((ext_vector_type(16))) _Float16 v16h;
typedef __attribute__((ext_vector_type(8)))  float    v8f;

// ---------------------------------------------------------------------------
// WMMA fragment layout (wave32, v_wmma_f32_16x16x32_f16):
//   A (16x32): lane m=l&15 holds row m; low lanes K {0-7,16-23}, high {8-15,24-31}
//   B (32x16): lane holds col n0+(l&15); low lanes K {0-7,16-23}, high {8-15,24-31}
//   C/D:       VGPR i -> row i (+8 for high lanes), col = n0 + (l&15)
// We stage operands in LDS pre-swizzled so each lane's 16 halves are one
// contiguous 32-byte chunk -> every fragment load is 2x ds_load_b128.
// ---------------------------------------------------------------------------

// dest index inside a swizzled 64x64 B-operand for source element W[k][n]
__device__ inline int bswz(int k, int n) {
    int kb = k >> 5, kp = k & 31;
    int sel = (kp >> 3) & 1, e = (kp & 7) + ((kp >> 4) << 3);
    return ((kb * 4 + (n >> 4)) * 32 + (n & 15) + sel * 16) * 16 + e;
}

// dest index inside a swizzled 16x64 A-tile for source element X[m][k]
__device__ inline int aswz(int m, int k) {
    int kb = k >> 5, kp = k & 31;
    int sel = (kp >> 3) & 1;
    int e = (kp & 7) + ((kp >> 4) << 3);
    return kb * 512 + (m + 16 * sel) * 16 + e;
}

__device__ inline v16h fragA_sw(const _Float16* __restrict__ xt, int kb) {
    int l = threadIdx.x & 31;
    return *reinterpret_cast<const v16h*>(xt + kb * 512 + l * 16);
}

// blk = kb*4 + j  (kb: K-block 0/1, j: 16-col tile 0..3)
__device__ inline v16h fragB_sw(const _Float16* __restrict__ Wsw, int blk) {
    int l = threadIdx.x & 31;
    return *reinterpret_cast<const v16h*>(Wsw + (blk * 32 + l) * 16);
}

// ---------------------------------------------------------------------------
// Utility kernels (float4-vectorized; all sizes here are multiples of 4)
// ---------------------------------------------------------------------------
__global__ void zero4_kernel(float4* __restrict__ p, long long n4) {
    long long i = (long long)blockIdx.x * blockDim.x + threadIdx.x;
    if (i < n4) p[i] = make_float4(0.f, 0.f, 0.f, 0.f);
}

__global__ void embed_kernel(const int* __restrict__ x, const float4* __restrict__ emb,
                             float4* __restrict__ h, int N) {
    long long i = (long long)blockIdx.x * blockDim.x + threadIdx.x;
    if (i >= (long long)N * 16) return;
    int n = (int)(i >> 4), cq = (int)(i & 15);
    h[i] = emb[x[n] * 16 + cq];
}

// ---------------------------------------------------------------------------
// Edge scatter:  agg[dst] += relu(h[src] + edge_attr)   (branch 1)
//                agg[dst] += h[src]                      (branch 2)
// ---------------------------------------------------------------------------
__global__ void scatter1_kernel(const float* __restrict__ h, const float* __restrict__ ea,
                                const int* __restrict__ src, const int* __restrict__ dst,
                                float* __restrict__ agg, int E) {
    long long i = (long long)blockIdx.x * blockDim.x + threadIdx.x;
    long long e = i >> 4;
    if (e >= E) return;
    int cg = (int)(i & 15) * 4;
    int s = src[e], d = dst[e];
    const float4 hv = *(const float4*)(h + (size_t)s * GDIM + cg);
    const float4 ev = *(const float4*)(ea + (size_t)e * GDIM + cg);
    float* ap = agg + (size_t)d * GDIM + cg;
    atomicAdd(ap + 0, fmaxf(hv.x + ev.x, 0.0f));
    atomicAdd(ap + 1, fmaxf(hv.y + ev.y, 0.0f));
    atomicAdd(ap + 2, fmaxf(hv.z + ev.z, 0.0f));
    atomicAdd(ap + 3, fmaxf(hv.w + ev.w, 0.0f));
}

__global__ void scatter2_kernel(const float* __restrict__ h,
                                const int* __restrict__ src, const int* __restrict__ dst,
                                float* __restrict__ agg, int E) {
    long long i = (long long)blockIdx.x * blockDim.x + threadIdx.x;
    long long e = i >> 4;
    if (e >= E) return;
    int cg = (int)(i & 15) * 4;
    int s = src[e], d = dst[e];
    const float4 hv = *(const float4*)(h + (size_t)s * GDIM + cg);
    float* ap = agg + (size_t)d * GDIM + cg;
    atomicAdd(ap + 0, hv.x);
    atomicAdd(ap + 1, hv.y);
    atomicAdd(ap + 2, hv.z);
    atomicAdd(ap + 3, hv.w);
}

// ---------------------------------------------------------------------------
// MLP:  t = relu((agg+h) @ W1 + b1) @ W2 + b2 ; accumulate BN column stats.
// 256 threads = 8 waves, each wave owns a 16-row x 64-col output tile.
// ---------------------------------------------------------------------------
__global__ __launch_bounds__(256) void mlp_kernel(
    const float* __restrict__ h, const float* __restrict__ agg,
    const float* __restrict__ W1, const float* __restrict__ b1,
    const float* __restrict__ W2, const float* __restrict__ b2,
    float* __restrict__ t, float* __restrict__ stats, int N) {
    __shared__ alignas(32) _Float16 sW1[GDIM * GDIM];   // swizzled
    __shared__ alignas(32) _Float16 sW2[GDIM * GDIM];   // swizzled
    __shared__ float sb1[GDIM], sb2[GDIM];
    __shared__ alignas(32) _Float16 sX[8][16 * GDIM];   // per-wave swizzled A tile
    __shared__ float sSum[GDIM], sSq[GDIM];

    int tid = threadIdx.x;
    for (int i = tid; i < GDIM * GDIM; i += 256) {
        int f = i >> 6, c = i & 63;
        int d = bswz(f, c);
        sW1[d] = (_Float16)W1[i];
        sW2[d] = (_Float16)W2[i];
    }
    if (tid < GDIM) { sb1[tid] = b1[tid]; sb2[tid] = b2[tid]; sSum[tid] = 0.f; sSq[tid] = 0.f; }
    __syncthreads();

    int wave = tid >> 5, lane = tid & 31;
    int row0 = blockIdx.x * 128 + wave * 16;
    _Float16* xt = sX[wave];

    // load x = agg + h tile (pad OOB rows with zeros), store swizzled
    for (int i = lane; i < 16 * GDIM; i += 32) {
        int m = i >> 6, k = i & 63;
        int r = row0 + m;
        float v = 0.0f;
        if (r < N) v = agg[(size_t)r * GDIM + k] + h[(size_t)r * GDIM + k];
        xt[aswz(m, k)] = (_Float16)v;
    }
    __syncthreads();

    // ---- GEMM 1: hidden = relu(x @ W1 + b1) ----
    v16h a0 = fragA_sw(xt, 0);
    v16h a1 = fragA_sw(xt, 1);
    v8f c1[4];
#pragma unroll
    for (int j = 0; j < 4; ++j) {
        v8f c = {};
        c = __builtin_amdgcn_wmma_f32_16x16x32_f16(false, a0, false, fragB_sw(sW1, j),
                                                   (short)0, c, false, false);
        c = __builtin_amdgcn_wmma_f32_16x16x32_f16(false, a1, false, fragB_sw(sW1, 4 + j),
                                                   (short)0, c, false, false);
        c1[j] = c;
    }
    // relu + bias, write hidden back into the wave's LDS tile (swizzled A layout)
    {
        int mh = (lane >> 4) ? 8 : 0;
#pragma unroll
        for (int j = 0; j < 4; ++j) {
            int col = j * 16 + (lane & 15);
#pragma unroll
            for (int i = 0; i < 8; ++i) {
                float v = fmaxf(c1[j][i] + sb1[col], 0.0f);
                xt[aswz(i + mh, col)] = (_Float16)v;
            }
        }
    }
    __syncthreads();

    // ---- GEMM 2: t = hidden @ W2 + b2 ----
    a0 = fragA_sw(xt, 0);
    a1 = fragA_sw(xt, 1);
    int mh = (lane >> 4) ? 8 : 0;
#pragma unroll
    for (int j = 0; j < 4; ++j) {
        v8f c = {};
        c = __builtin_amdgcn_wmma_f32_16x16x32_f16(false, a0, false, fragB_sw(sW2, j),
                                                   (short)0, c, false, false);
        c = __builtin_amdgcn_wmma_f32_16x16x32_f16(false, a1, false, fragB_sw(sW2, 4 + j),
                                                   (short)0, c, false, false);
        int col = j * 16 + (lane & 15);
#pragma unroll
        for (int i = 0; i < 8; ++i) {
            int r = row0 + i + mh;
            if (r < N) {
                float v = c[i] + sb2[col];
                t[(size_t)r * GDIM + col] = v;
                atomicAdd(&sSum[col], v);        // ds_add_f32
                atomicAdd(&sSq[col], v * v);
            }
        }
    }
    __syncthreads();
    if (tid < GDIM) {
        atomicAdd(&stats[tid], sSum[tid]);
        atomicAdd(&stats[GDIM + tid], sSq[tid]);
    }
}

// h += gamma * (t - mean) * rsqrt(var + 1e-5) + beta   (biased batch stats)
__global__ void bnapply_kernel(float4* __restrict__ h, const float4* __restrict__ t,
                               const float4* __restrict__ stats,
                               const float4* __restrict__ gamma, const float4* __restrict__ beta,
                               int N) {
    long long i = (long long)blockIdx.x * blockDim.x + threadIdx.x;
    if (i >= (long long)N * 16) return;
    int cq = (int)(i & 15);
    float invN = 1.0f / (float)N;
    float4 sm = stats[cq];          // column sums
    float4 sq = stats[16 + cq];     // column sums of squares
    float4 g = gamma[cq], be = beta[cq], tv = t[i], hv = h[i];
    float m, var;
    m = sm.x * invN; var = sq.x * invN - m * m;
    hv.x += g.x * (tv.x - m) * rsqrtf(var + 1e-5f) + be.x;
    m = sm.y * invN; var = sq.y * invN - m * m;
    hv.y += g.y * (tv.y - m) * rsqrtf(var + 1e-5f) + be.y;
    m = sm.z * invN; var = sq.z * invN - m * m;
    hv.z += g.z * (tv.z - m) * rsqrtf(var + 1e-5f) + be.z;
    m = sm.w * invN; var = sq.w * invN - m * m;
    hv.w += g.w * (tv.w - m) * rsqrtf(var + 1e-5f) + be.w;
    h[i] = hv;
}

// ---------------------------------------------------------------------------
// Attention projection:  out[n, h*8+d] = sum_f feat[n,f] * W[h][f][d]
// feat = [x (64 cols), direction] -> WMMA over K=64 plus rank-1 direction term.
// ---------------------------------------------------------------------------
__global__ __launch_bounds__(256) void proj_kernel(
    const float* __restrict__ X, const float* __restrict__ dir,
    const float* __restrict__ W,   // [8][65][8]
    float* __restrict__ out, int N) {
    __shared__ alignas(32) _Float16 sW[GDIM * GDIM];   // swizzled; col c = h*8+d
    __shared__ float sWl[GDIM];                        // direction row
    __shared__ alignas(32) _Float16 sX[8][16 * GDIM];
    __shared__ float sDir[8][16];

    int tid = threadIdx.x;
    for (int i = tid; i < GDIM * GDIM; i += 256) {
        int f = i >> 6, c = i & 63;
        sW[bswz(f, c)] = (_Float16)W[(c >> 3) * 520 + f * 8 + (c & 7)];
    }
    if (tid < GDIM) sWl[tid] = W[(tid >> 3) * 520 + 512 + (tid & 7)];
    __syncthreads();

    int wave = tid >> 5, lane = tid & 31;
    int row0 = blockIdx.x * 128 + wave * 16;
    _Float16* xt = sX[wave];
    for (int i = lane; i < 16 * GDIM; i += 32) {
        int m = i >> 6, k = i & 63;
        int r = row0 + m;
        xt[aswz(m, k)] = (_Float16)((r < N) ? X[(size_t)r * GDIM + k] : 0.0f);
    }
    if (lane < 16) {
        int r = row0 + lane;
        sDir[wave][lane] = (r < N) ? dir[r] : 0.0f;
    }
    __syncthreads();

    v16h a0 = fragA_sw(xt, 0);
    v16h a1 = fragA_sw(xt, 1);
    int mh = (lane >> 4) ? 8 : 0;
#pragma unroll
    for (int j = 0; j < 4; ++j) {
        v8f c = {};
        c = __builtin_amdgcn_wmma_f32_16x16x32_f16(false, a0, false, fragB_sw(sW, j),
                                                   (short)0, c, false, false);
        c = __builtin_amdgcn_wmma_f32_16x16x32_f16(false, a1, false, fragB_sw(sW, 4 + j),
                                                   (short)0, c, false, false);
        int col = j * 16 + (lane & 15);
#pragma unroll
        for (int i = 0; i < 8; ++i) {
            int m = i + mh;
            int r = row0 + m;
            if (r < N) out[(size_t)r * GDIM + col] = c[i] + sDir[wave][m] * sWl[col];
        }
    }
}

// K-vectors for the B target nodes:  Kout[b, h*8+d] = feat(target[b]) . Wk[h][:,d]
__global__ void kproj_kernel(const float* __restrict__ h, const float* __restrict__ dir,
                             const int* __restrict__ tgt, const float* __restrict__ Wk,
                             float* __restrict__ Kout, int B) {
    int i = blockIdx.x * blockDim.x + threadIdx.x;
    if (i >= B * GDIM) return;
    int b = i >> 6, c = i & 63;
    int n = tgt[b];
    const float* w = Wk + (c >> 3) * 520 + (c & 7);
    float s = 0.0f;
#pragma unroll
    for (int f = 0; f < GDIM; ++f) s += h[(size_t)n * GDIM + f] * w[f * 8];
    s += dir[n] * w[512];
    Kout[i] = s;
}

// logits l{1,2}[n,h] = scale * Q[n,h,:] . K{1,2}[batch[n],h,:]
__global__ void logits_kernel(const float* __restrict__ Q, const float* __restrict__ K1,
                              const float* __restrict__ K2, const int* __restrict__ batch,
                              float* __restrict__ l1, float* __restrict__ l2, int N) {
    long long i = (long long)blockIdx.x * blockDim.x + threadIdx.x;
    if (i >= (long long)N * GHEAD) return;
    int n = (int)(i >> 3), hd = (int)(i & 7);
    int b = batch[n];
    const float4* q4 = (const float4*)(Q + (size_t)n * GDIM + hd * 8);
    const float4* k14 = (const float4*)(K1 + b * GDIM + hd * 8);
    const float4* k24 = (const float4*)(K2 + b * GDIM + hd * 8);
    float4 qa = q4[0], qb = q4[1];
    float4 ka = k14[0], kb = k14[1];
    float4 kc = k24[0], kd = k24[1];
    float s1 = qa.x * ka.x + qa.y * ka.y + qa.z * ka.z + qa.w * ka.w +
               qb.x * kb.x + qb.y * kb.y + qb.z * kb.z + qb.w * kb.w;
    float s2 = qa.x * kc.x + qa.y * kc.y + qa.z * kc.z + qa.w * kc.w +
               qb.x * kd.x + qb.y * kd.y + qb.z * kd.z + qb.w * kd.w;
    const float scale = 0.35355339059327373f;  // 1/sqrt(8)
    l1[i] = s1 * scale;
    l2[i] = s2 * scale;
}

// per-graph node ranges (batch is sorted)
__global__ void starts_kernel(const int* __restrict__ batch, int* __restrict__ starts,
                              int N, int B) {
    int n = blockIdx.x * blockDim.x + threadIdx.x;
    if (n >= N) return;
    int b = batch[n];
    int bp = (n == 0) ? -1 : batch[n - 1];
    for (int g = bp + 1; g <= b; ++g) starts[g] = n;
    if (n == N - 1)
        for (int g = b + 1; g <= B; ++g) starts[g] = N;
}

// ---------------------------------------------------------------------------
// Fused per-graph segment-softmax (two targets) + weighted pool of V.
// One block (256 threads) per graph.  o[b,:] += pooled  (o pre-zeroed).
// ---------------------------------------------------------------------------
__global__ __launch_bounds__(256) void attnpool_kernel(
    const float* __restrict__ V, const float* __restrict__ l1, const float* __restrict__ l2,
    const int* __restrict__ starts, float* __restrict__ o) {
    int b = blockIdx.x;
    int s = starts[b], e = starts[b + 1];
    __shared__ float red[8][16];
    __shared__ float m1s[GHEAD], m2s[GHEAD], d1s[GHEAD], d2s[GHEAD];
    __shared__ float acc[4][GDIM];
    int tid = threadIdx.x, wave = tid >> 5, lane = tid & 31;

    // phase 1: per-head max of l1, l2
    float mx[16];
#pragma unroll
    for (int j = 0; j < 16; ++j) mx[j] = -1e30f;
    for (int n = s + tid; n < e; n += 256) {
        const float4* p1 = (const float4*)(l1 + (size_t)n * 8);
        const float4* p2 = (const float4*)(l2 + (size_t)n * 8);
        float4 a = p1[0], bq = p1[1], c = p2[0], d = p2[1];
        mx[0] = fmaxf(mx[0], a.x);  mx[1] = fmaxf(mx[1], a.y);
        mx[2] = fmaxf(mx[2], a.z);  mx[3] = fmaxf(mx[3], a.w);
        mx[4] = fmaxf(mx[4], bq.x); mx[5] = fmaxf(mx[5], bq.y);
        mx[6] = fmaxf(mx[6], bq.z); mx[7] = fmaxf(mx[7], bq.w);
        mx[8] = fmaxf(mx[8], c.x);  mx[9] = fmaxf(mx[9], c.y);
        mx[10] = fmaxf(mx[10], c.z); mx[11] = fmaxf(mx[11], c.w);
        mx[12] = fmaxf(mx[12], d.x); mx[13] = fmaxf(mx[13], d.y);
        mx[14] = fmaxf(mx[14], d.z); mx[15] = fmaxf(mx[15], d.w);
    }
#pragma unroll
    for (int off = 16; off; off >>= 1)
#pragma unroll
        for (int j = 0; j < 16; ++j) mx[j] = fmaxf(mx[j], __shfl_xor(mx[j], off, 32));
    if (lane < 16) red[wave][lane] = mx[lane];
    __syncthreads();
    if (tid < 16) {
        float v = red[0][tid];
        for (int w = 1; w < 8; ++w) v = fmaxf(v, red[w][tid]);
        if (tid < 8) m1s[tid] = v; else m2s[tid - 8] = v;
    }
    __syncthreads();

    // phase 2: per-head sum of exp
    float sm[16];
#pragma unroll
    for (int j = 0; j < 16; ++j) sm[j] = 0.0f;
    for (int n = s + tid; n < e; n += 256) {
        const float4* p1 = (const float4*)(l1 + (size_t)n * 8);
        const float4* p2 = (const float4*)(l2 + (size_t)n * 8);
        float4 a = p1[0], bq = p1[1], c = p2[0], d = p2[1];
        sm[0] += __expf(a.x - m1s[0]);  sm[1] += __expf(a.y - m1s[1]);
        sm[2] += __expf(a.z - m1s[2]);  sm[3] += __expf(a.w - m1s[3]);
        sm[4] += __expf(bq.x - m1s[4]); sm[5] += __expf(bq.y - m1s[5]);
        sm[6] += __expf(bq.z - m1s[6]); sm[7] += __expf(bq.w - m1s[7]);
        sm[8] += __expf(c.x - m2s[0]);  sm[9] += __expf(c.y - m2s[1]);
        sm[10] += __expf(c.z - m2s[2]); sm[11] += __expf(c.w - m2s[3]);
        sm[12] += __expf(d.x - m2s[4]); sm[13] += __expf(d.y - m2s[5]);
        sm[14] += __expf(d.z - m2s[6]); sm[15] += __expf(d.w - m2s[7]);
    }
#pragma unroll
    for (int off = 16; off; off >>= 1)
#pragma unroll
        for (int j = 0; j < 16; ++j) sm[j] += __shfl_xor(sm[j], off, 32);
    if (lane < 16) red[wave][lane] = sm[lane];
    __syncthreads();
    if (tid < 16) {
        float v = 0.0f;
        for (int w = 0; w < 8; ++w) v += red[w][tid];
        v = 1.0f / (v + 1e-16f);
        if (tid < 8) d1s[tid] = v; else d2s[tid - 8] = v;
    }
    __syncthreads();

    // phase 3: pooled[col] = sum_n (s1+s2)[n, col>>3] * V[n, col]
    int col = tid & 63, grp = tid >> 6, hd = col >> 3;
    float m1 = m1s[hd], m2 = m2s[hd], id1 = d1s[hd], id2 = d2s[hd];
    float a = 0.0f;
    for (int n = s + grp; n < e; n += 4) {
        float w = __expf(l1[(size_t)n * 8 + hd] - m1) * id1 +
                  __expf(l2[(size_t)n * 8 + hd] - m2) * id2;
        a += w * V[(size_t)n * GDIM + col];
    }
    acc[grp][col] = a;
    __syncthreads();
    if (tid < GDIM) {
        float v = acc[0][tid] + acc[1][tid] + acc[2][tid] + acc[3][tid];
        o[b * GDIM + tid] += v;
    }
}

// final head: sigmoid(gelu([o, dir2] @ fc11 + b) @ fc12 + b)
__global__ __launch_bounds__(32) void head_kernel(
    const float* __restrict__ o, const float* __restrict__ dir2,
    const float* __restrict__ fc11w, const float* __restrict__ fc11b,
    const float* __restrict__ fc12w, const float* __restrict__ fc12b,
    float* __restrict__ out) {
    int b = blockIdx.x;
    __shared__ float hid[32];
    int j = threadIdx.x;
    if (j < 32) {
        float s = fc11b[j];
        for (int f = 0; f < GDIM; ++f) s += o[b * GDIM + f] * fc11w[f * 32 + j];
        s += dir2[b] * fc11w[GDIM * 32 + j];
        hid[j] = 0.5f * s * (1.0f + erff(s * 0.70710678118654752f));  // exact gelu
    }
    __syncthreads();
    if (j == 0) {
        float s = fc12b[0];
        for (int k = 0; k < 32; ++k) s += hid[k] * fc12w[k];
        out[b] = 1.0f / (1.0f + __expf(-s));
    }
}

// ---------------------------------------------------------------------------
extern "C" void kernel_launch(void* const* d_in, const int* in_sizes, int n_in,
                              void* d_out, int out_size, void* d_ws, size_t ws_size,
                              hipStream_t stream) {
    (void)n_in; (void)out_size; (void)ws_size;
    const int N = in_sizes[0];
    const int E = in_sizes[2] / 2;
    const int B = in_sizes[9];

    const int*   x1    = (const int*)d_in[0];
    const int*   x2    = (const int*)d_in[1];
    const int*   ei1   = (const int*)d_in[2];
    const float* ea1   = (const float*)d_in[3];
    const int*   ei2   = (const int*)d_in[4];
    const int*   batch = (const int*)d_in[5];
    const int*   tgt1  = (const int*)d_in[6];
    const int*   tgt2  = (const int*)d_in[7];
    const float* dir   = (const float*)d_in[8];
    const float* dir2  = (const float*)d_in[9];
    const float* emb1  = (const float*)d_in[10];
    const float* emb2  = (const float*)d_in[11];
    const float* m1w1  = (const float*)d_in[12];
    const float* m1b1  = (const float*)d_in[13];
    const float* m1w2  = (const float*)d_in[14];
    const float* m1b2  = (const float*)d_in[15];
    const float* g1    = (const float*)d_in[16];
    const float* be1   = (const float*)d_in[17];
    const float* Wq1   = (const float*)d_in[18];
    const float* Wk1   = (const float*)d_in[19];
    const float* Wv1   = (const float*)d_in[20];
    const float* m2w1  = (const float*)d_in[21];
    const float* m2b1  = (const float*)d_in[22];
    const float* m2w2  = (const float*)d_in[23];
    const float* m2b2  = (const float*)d_in[24];
    const float* g2    = (const float*)d_in[25];
    const float* be2   = (const float*)d_in[26];
    const float* Wq2   = (const float*)d_in[27];
    const float* Wk2   = (const float*)d_in[28];
    const float* Wv2   = (const float*)d_in[29];
    const float* fc11w = (const float*)d_in[30];
    const float* fc11b = (const float*)d_in[31];
    const float* fc12w = (const float*)d_in[32];
    const float* fc12b = (const float*)d_in[33];

    const int* src1 = ei1, *dst1 = ei1 + E;
    const int* src2 = ei2, *dst2 = ei2 + E;

    // workspace layout
    char* ws = (char*)d_ws;
    size_t nd = (size_t)N * GDIM;
    float* h1    = (float*)ws; ws += nd * sizeof(float);
    float* h2    = (float*)ws; ws += nd * sizeof(float);
    float* agg   = (float*)ws; ws += nd * sizeof(float);   // later reused as Q
    float* t     = (float*)ws; ws += nd * sizeof(float);   // later reused as V
    float* l1    = (float*)ws; ws += (size_t)N * 8 * sizeof(float);
    float* l2    = (float*)ws; ws += (size_t)N * 8 * sizeof(float);
    float* K1    = (float*)ws; ws += (size_t)B * GDIM * sizeof(float);
    float* K2    = (float*)ws; ws += (size_t)B * GDIM * sizeof(float);
    float* o     = (float*)ws; ws += (size_t)B * GDIM * sizeof(float);
    float* stats = (float*)ws; ws += 2 * GDIM * sizeof(float);
    int*   starts = (int*)ws;  ws += (size_t)(B + 1) * sizeof(int);

    const int TB = 256;
    long long n4 = (long long)N * 16;             // N*64 floats as float4
    int gN4    = (int)((n4 + TB - 1) / TB);
    int gE     = (int)(((long long)E * 16 + TB - 1) / TB);
    int gMLP   = (N + 127) / 128;
    int gLog   = (int)(((long long)N * 8 + TB - 1) / TB);
    int gN     = (N + TB - 1) / TB;
    int gK     = (B * GDIM + TB - 1) / TB;

    // embeddings
    embed_kernel<<<gN4, TB, 0, stream>>>(x1, (const float4*)emb1, (float4*)h1, N);
    embed_kernel<<<gN4, TB, 0, stream>>>(x2, (const float4*)emb2, (float4*)h2, N);

    // branch 1: GINEConv stack
    for (int i = 0; i < 4; ++i) {
        zero4_kernel<<<gN4, TB, 0, stream>>>((float4*)agg, n4);
        zero4_kernel<<<1, TB, 0, stream>>>((float4*)stats, 32);
        scatter1_kernel<<<gE, TB, 0, stream>>>(h1, ea1, src1, dst1, agg, E);
        mlp_kernel<<<gMLP, TB, 0, stream>>>(h1, agg,
                                            m1w1 + (size_t)i * GDIM * GDIM, m1b1 + i * GDIM,
                                            m1w2 + (size_t)i * GDIM * GDIM, m1b2 + i * GDIM,
                                            t, stats, N);
        bnapply_kernel<<<gN4, TB, 0, stream>>>((float4*)h1, (const float4*)t,
                                               (const float4*)stats,
                                               (const float4*)(g1 + i * GDIM),
                                               (const float4*)(be1 + i * GDIM), N);
    }
    // branch 2: GINConv stack
    for (int i = 0; i < 4; ++i) {
        zero4_kernel<<<gN4, TB, 0, stream>>>((float4*)agg, n4);
        zero4_kernel<<<1, TB, 0, stream>>>((float4*)stats, 32);
        scatter2_kernel<<<gE, TB, 0, stream>>>(h2, src2, dst2, agg, E);
        mlp_kernel<<<gMLP, TB, 0, stream>>>(h2, agg,
                                            m2w1 + (size_t)i * GDIM * GDIM, m2b1 + i * GDIM,
                                            m2w2 + (size_t)i * GDIM * GDIM, m2b2 + i * GDIM,
                                            t, stats, N);
        bnapply_kernel<<<gN4, TB, 0, stream>>>((float4*)h2, (const float4*)t,
                                               (const float4*)stats,
                                               (const float4*)(g2 + i * GDIM),
                                               (const float4*)(be2 + i * GDIM), N);
    }

    // attention pooling (agg := Q, t := V from here on)
    zero4_kernel<<<(B * 16 + TB - 1) / TB, TB, 0, stream>>>((float4*)o, B * 16);
    starts_kernel<<<gN, TB, 0, stream>>>(batch, starts, N, B);

    // branch 1
    kproj_kernel<<<gK, TB, 0, stream>>>(h1, dir, tgt1, Wk1, K1, B);
    kproj_kernel<<<gK, TB, 0, stream>>>(h1, dir, tgt2, Wk1, K2, B);
    proj_kernel<<<gMLP, TB, 0, stream>>>(h1, dir, Wq1, agg, N);
    proj_kernel<<<gMLP, TB, 0, stream>>>(h1, dir, Wv1, t, N);
    logits_kernel<<<gLog, TB, 0, stream>>>(agg, K1, K2, batch, l1, l2, N);
    attnpool_kernel<<<B, TB, 0, stream>>>(t, l1, l2, starts, o);

    // branch 2
    kproj_kernel<<<gK, TB, 0, stream>>>(h2, dir, tgt1, Wk2, K1, B);
    kproj_kernel<<<gK, TB, 0, stream>>>(h2, dir, tgt2, Wk2, K2, B);
    proj_kernel<<<gMLP, TB, 0, stream>>>(h2, dir, Wq2, agg, N);
    proj_kernel<<<gMLP, TB, 0, stream>>>(h2, dir, Wv2, t, N);
    logits_kernel<<<gLog, TB, 0, stream>>>(agg, K1, K2, batch, l1, l2, N);
    attnpool_kernel<<<B, TB, 0, stream>>>(t, l1, l2, starts, o);

    // final head
    head_kernel<<<B, 32, 0, stream>>>(o, dir2, fc11w, fc11b, fc12w, fc12b, (float*)d_out);
}
